// RNNModel_8632884265028
// MI455X (gfx1250) — matrix operand order repaired and checked by reading, not compiled, not run on hardware
//
#include <hip/hip_runtime.h>
#include <cstdint>
#include <cstddef>

// ---------------------------------------------------------------------------
// Types for CDNA5 WMMA (gfx1250, wave32)
// ---------------------------------------------------------------------------
typedef __bf16 bf16;
typedef __attribute__((ext_vector_type(16))) __bf16 v16bf;
typedef __attribute__((ext_vector_type(8)))  __bf16 v8bf;
typedef __attribute__((ext_vector_type(4)))  __bf16 v4bf;
typedef __attribute__((ext_vector_type(8)))  float  v8f;
typedef __attribute__((ext_vector_type(4)))  unsigned int v4u;

union FragBF { v16bf v; v4u u[2]; };

__device__ __forceinline__ float sigmoidf_(float x) { return 1.0f / (1.0f + expf(-x)); }

// Async copy of one 16B chunk: global -> LDS (CDNA5 async-to-LDS path, ASYNCcnt).
__device__ __forceinline__ void async_g2l_b128(unsigned ldsaddr, const void* gptr) {
  asm volatile("global_load_async_to_lds_b128 %0, %1, off"
               :: "v"(ldsaddr), "v"(gptr) : "memory");
}

// ---------------------------------------------------------------------------
// WMMA GEMM:  C[M,N] (+)= A[M,K] @ B[N,K]^T     (A,B bf16 row-major, K-contig)
// MODE 0: C = A*B^T + bias (f32)
// MODE 1: C += A*B^T       (f32 accumulate, no bias)
// MODE 2: C = bf16(tanh(A*B^T + bias))   (C is bf16*)
// MODE 4: C = A*B^T + bias (f32, non-temporal stores — streaming output)
//
// Block = 128 threads = 4 waves. Wave tile = 32 (2 m-tiles) x 64 (4 n-tiles).
// B tile (64 rows x 32 k) staged in LDS via async global->LDS b128 copies,
// ping-pong double-buffered; A fragments use two disjoint register sets with
// the k-loop manually unrolled x2 (no rotation copies / hazard NOPs).
// ksplit: split-K across blockIdx.z, partials at C + split*M*N.
// Requires: (K/ksplit) % 32 == 0, N % 64 == 0, M % 32 == 0.
// ---------------------------------------------------------------------------
#define MT 2
#define NT 4
#define KSTEP 32
#define BLOCK_ROWS 128
#define LDS_PITCH 80                    // 64 data bytes + 16 pad per B-tile row
#define LDS_BUF  (64 * LDS_PITCH)       // one B-tile buffer

template <int MODE>
__global__ __launch_bounds__(128) void wmma_gemm_bf16_kernel(
    const bf16* __restrict__ A, const bf16* __restrict__ Bm, float* __restrict__ C,
    int M, int N, int K,
    long long strideA, long long strideB, long long strideC,
    const float* __restrict__ bias, int ksplit)
{
  __shared__ alignas(16) char smem[2 * LDS_BUF];

  const int tid  = threadIdx.x;
  const int lane = tid & 31;
  const int wave = tid >> 5;
  const int half = lane >> 4;
  const int lh   = lane & 15;

  const int split = blockIdx.z % ksplit;
  const int batch = blockIdx.z / ksplit;
  const int Ks = K / ksplit;

  const bf16* Ab = A  + (long long)batch * strideA + split * Ks;
  const bf16* Bb = Bm + (long long)batch * strideB + split * Ks;
  float*      Cb = C  + (long long)batch * strideC + (size_t)split * M * N;

  const int n0 = blockIdx.x * 64;
  const int m0 = blockIdx.y * BLOCK_ROWS + wave * (MT * 16);
  const bool active = (m0 < M);           // wave-uniform; EXEC all-1 inside

  // --- B staging plan: 256 16B chunks per 64x32 tile, 2 chunks per thread ---
  const int c0 = tid * 2;
  const int brow0 = c0 >> 2,       bk0 = c0 & 3;
  const int brow1 = (c0 + 1) >> 2, bk1 = (c0 + 1) & 3;
  const bf16* bsrc0 = Bb + (size_t)(n0 + brow0) * K + bk0 * 8;
  const bf16* bsrc1 = Bb + (size_t)(n0 + brow1) * K + bk1 * 8;
  const unsigned lbase = (unsigned)(uintptr_t)&smem[0];   // LDS addr = low 32 bits
  const unsigned ldst0 = lbase + brow0 * LDS_PITCH + bk0 * 16;
  const unsigned ldst1 = lbase + brow1 * LDS_PITCH + bk1 * 16;

  // --- A fragment pointers (ISA 7.12.2 16-bit A layout) ---
  const bf16* Ap[MT];
#pragma unroll
  for (int mt = 0; mt < MT; ++mt)
    Ap[mt] = Ab + (size_t)(m0 + mt * 16 + lh) * K;

  v8f acc[MT][NT];
#pragma unroll
  for (int mt = 0; mt < MT; ++mt)
#pragma unroll
    for (int nt = 0; nt < NT; ++nt)
#pragma unroll
      for (int r = 0; r < 8; ++r) acc[mt][nt][r] = 0.0f;

  FragBF fa0[MT], fa1[MT];

  // helper lambdas (device, inlined)
  auto loadA = [&](FragBF (&fa)[MT], int k0) {
#pragma unroll
    for (int mt = 0; mt < MT; ++mt) {
      fa[mt].u[0] = *(const v4u*)(Ap[mt] + k0 + half * 8);
      fa[mt].u[1] = *(const v4u*)(Ap[mt] + k0 + 16 + half * 8);
    }
  };
  auto stageB = [&](int buf, int k0) {
    const unsigned boff = buf * LDS_BUF;
    async_g2l_b128(ldst0 + boff, bsrc0 + k0);
    async_g2l_b128(ldst1 + boff, bsrc1 + k0);
  };
  auto mma_step = [&](int buf, FragBF (&fa)[MT]) {
    const char* bbuf = smem + buf * LDS_BUF;
    FragBF fb[NT];
#pragma unroll
    for (int nt = 0; nt < NT; ++nt) {   // all LDS reads first: one dscnt wait
      const char* brow = bbuf + (nt * 16 + lh) * LDS_PITCH + half * 16;
      fb[nt].u[0] = *(const v4u*)(brow);
      fb[nt].u[1] = *(const v4u*)(brow + 32);
    }
#pragma unroll
    for (int nt = 0; nt < NT; ++nt)     // 8 back-to-back WMMAs
#pragma unroll
      for (int mt = 0; mt < MT; ++mt)
        acc[mt][nt] = __builtin_amdgcn_wmma_f32_16x16x32_bf16(
            false, fa[mt].v, false, fb[nt].v, (short)0, acc[mt][nt], false, false);
  };

  const int nk = Ks / KSTEP;
  // ---- prologue: stage k-step 0 into buf0 / fa0 ----
  stageB(0, 0);
  if (active) loadA(fa0, 0);

  for (int kk = 0; kk < nk; kk += 2) {
    // ---- half A: consume buf0/fa0, prefetch (kk+1) into buf1/fa1 ----
    {
      const bool hasnext = (kk + 1 < nk);
      if (hasnext) {
        stageB(1, (kk + 1) * KSTEP);
        if (active) loadA(fa1, (kk + 1) * KSTEP);
        asm volatile("s_wait_asynccnt 0x2" ::: "memory");
      } else {
        asm volatile("s_wait_asynccnt 0x0" ::: "memory");
      }
      __syncthreads();
      if (active) mma_step(0, fa0);
      __syncthreads();
      if (!hasnext) break;
    }
    // ---- half B: consume buf1/fa1, prefetch (kk+2) into buf0/fa0 ----
    {
      const bool hasnext = (kk + 2 < nk);
      if (hasnext) {
        stageB(0, (kk + 2) * KSTEP);
        if (active) loadA(fa0, (kk + 2) * KSTEP);
        asm volatile("s_wait_asynccnt 0x2" ::: "memory");
      } else {
        asm volatile("s_wait_asynccnt 0x0" ::: "memory");
      }
      __syncthreads();
      if (active) mma_step(1, fa1);
      __syncthreads();
    }
  }

  if (!active) return;
  // D layout: VGPR r -> row mtile*16 + half*8 + r, column n0 + nt*16 + lh
#pragma unroll
  for (int mt = 0; mt < MT; ++mt) {
#pragma unroll
    for (int nt = 0; nt < NT; ++nt) {
      const int ncol = n0 + nt * 16 + lh;
      const float bv = (bias != nullptr && MODE != 1) ? bias[ncol] : 0.0f;
#pragma unroll
      for (int r = 0; r < 8; ++r) {
        const int row = m0 + mt * 16 + half * 8 + r;
        const size_t idx = (size_t)row * N + ncol;
        const float v = acc[mt][nt][r];
        if (MODE == 0)      Cb[idx] = v + bv;
        else if (MODE == 1) Cb[idx] += v;
        else if (MODE == 2) ((bf16*)Cb)[idx] = (bf16)tanhf(v + bv);
        else                __builtin_nontemporal_store(v + bv, &Cb[idx]);
      }
    }
  }
}

// ---------------------------------------------------------------------------
// Elementwise / gather / reduce helper kernels
// ---------------------------------------------------------------------------
__global__ void cvt_bf16_kernel(const float* __restrict__ s, bf16* __restrict__ d, size_t n) {
  size_t i = ((size_t)blockIdx.x * blockDim.x + threadIdx.x) * 8;
  const size_t stride = (size_t)gridDim.x * blockDim.x * 8;
  for (; i < n; i += stride) {
    const float4 f0 = *(const float4*)(s + i);
    const float4 f1 = *(const float4*)(s + i + 4);
    v8bf o;
    o[0] = (bf16)f0.x; o[1] = (bf16)f0.y; o[2] = (bf16)f0.z; o[3] = (bf16)f0.w;
    o[4] = (bf16)f1.x; o[5] = (bf16)f1.y; o[6] = (bf16)f1.z; o[7] = (bf16)f1.w;
    *(v8bf*)(d + i) = o;
  }
}

// X[t*B+b][e] = bf16(emb[tok][e]); 256 threads x 4 elems, b128 in / b64 out
__global__ void embed_kernel(const int* __restrict__ ids, const float* __restrict__ emb,
                             bf16* __restrict__ X, int T, int B, int E, int reverse) {
  const int tb = blockIdx.x;
  const int t = tb / B, b = tb % B;
  const int st = reverse ? (T - 1 - t) : t;
  const int tok = ids[st * B + b];
  const float* e = emb + (size_t)tok * E;
  bf16* x = X + (size_t)tb * E;
  const int j = threadIdx.x * 4;
  const float4 f = *(const float4*)(e + j);
  v4bf o; o[0] = (bf16)f.x; o[1] = (bf16)f.y; o[2] = (bf16)f.z; o[3] = (bf16)f.w;
  *(v4bf*)(x + j) = o;
}

__global__ void init_state_kernel(const float* __restrict__ h0, const float* __restrict__ c0,
                                  bf16* __restrict__ h_bf, float* __restrict__ c_st, int n) {
  const int i = blockIdx.x * blockDim.x + threadIdx.x;
  if (i < n) { h_bf[i] = (bf16)h0[i]; c_st[i] = c0[i]; }
}

// Fused LSTM cell: gates = G[t] + sum_p hW_p + bih + bhh ; update c,h
__global__ void lstm_cell_kernel(const float* __restrict__ G,   // (B,4H) slice for step t
                                 const float* __restrict__ hW,  // (nparts,B,4H) split-K partials
                                 const float* __restrict__ bih, const float* __restrict__ bhh,
                                 float* __restrict__ c_st, bf16* __restrict__ h_bf,
                                 float* __restrict__ out_f32,   // (T,B,H) base or nullptr
                                 bf16* __restrict__ x_next,     // (T*B,H) base or nullptr
                                 int B, int H, int step_t, int out_t, int nparts) {
  const int idx = blockIdx.x * blockDim.x + threadIdx.x;   // over B*H
  if (idx >= B * H) return;
  const int b = idx / H, j = idx % H;
  const float* g = G + (size_t)b * 4 * H;
  float ig = g[j]         + bih[j]         + bhh[j];
  float fg = g[H + j]     + bih[H + j]     + bhh[H + j];
  float gg = g[2 * H + j] + bih[2 * H + j] + bhh[2 * H + j];
  float og = g[3 * H + j] + bih[3 * H + j] + bhh[3 * H + j];
  const size_t P = (size_t)B * 4 * H;
  for (int p = 0; p < nparts; ++p) {
    const float* hw = hW + p * P + (size_t)b * 4 * H;
    ig += hw[j]; fg += hw[H + j]; gg += hw[2 * H + j]; og += hw[3 * H + j];
  }
  const float c = sigmoidf_(fg) * c_st[idx] + sigmoidf_(ig) * tanhf(gg);
  const float h = sigmoidf_(og) * tanhf(c);
  c_st[idx] = c;
  h_bf[idx] = (bf16)h;
  if (out_f32) out_f32[((size_t)out_t * B + b) * H + j] = h;
  if (x_next)  x_next[((size_t)step_t * B + b) * H + j] = (bf16)h;
}

// a = outL.w1 + outR.w2 + b ; sL = outL.w3 ; sR = outR.w3  (one block per (t,b))
__global__ void attn_dots_kernel(const float* __restrict__ outL, const float* __restrict__ outR,
                                 const float* __restrict__ attn_w, const float* __restrict__ attn_b,
                                 float* __restrict__ a_s, float* __restrict__ sL_s,
                                 float* __restrict__ sR_s, int H) {
  const int tb = blockIdx.x;
  const float* xl = outL + (size_t)tb * H;
  const float* xr = outR + (size_t)tb * H;
  const float* w1 = attn_w;
  const float* w2 = attn_w + H;
  const float* w3 = attn_w + 2 * H;
  float s1 = 0, s2 = 0, s3 = 0, s4 = 0;
  for (int j = threadIdx.x; j < H; j += blockDim.x) {
    const float l = xl[j], r = xr[j];
    s1 += l * w1[j]; s2 += r * w2[j]; s3 += l * w3[j]; s4 += r * w3[j];
  }
  __shared__ float r1[256], r2[256], r3[256], r4[256];
  const int tid = threadIdx.x;
  r1[tid] = s1; r2[tid] = s2; r3[tid] = s3; r4[tid] = s4;
  __syncthreads();
  for (int s = 128; s > 0; s >>= 1) {
    if (tid < s) { r1[tid] += r1[tid+s]; r2[tid] += r2[tid+s]; r3[tid] += r3[tid+s]; r4[tid] += r4[tid+s]; }
    __syncthreads();
  }
  if (tid == 0) { a_s[tb] = r1[0] + r2[0] + attn_b[0]; sL_s[tb] = r3[0]; sR_s[tb] = r4[0]; }
}

// Per (i,b): scores[k]=tanh(a + ctx[k]) over k=0..T, softmax over k,
// emit masked bf16 rows Pl[b][i][k]=p[k]*(k<=i), Pr[b][i][k]=p[k+1]*(k+1>i)
__global__ void attn_softmax_kernel(const float* __restrict__ a_s, const float* __restrict__ sL_s,
                                    const float* __restrict__ sR_s,
                                    bf16* __restrict__ Pl, bf16* __restrict__ Pr, int T, int B) {
  const int i = blockIdx.x, b = blockIdx.y, tid = threadIdx.x;
  __shared__ float sc[257];
  __shared__ float red[256];
  const float av = a_s[i * B + b];
  for (int k = tid; k <= T; k += 256) {
    const float ctx = (k <= i) ? sL_s[k * B + b] : sR_s[(k - 1) * B + b];
    sc[k] = tanhf(av + ctx);
  }
  __syncthreads();
  float m = sc[tid];
  if (tid == 0) m = fmaxf(m, sc[256]);
  red[tid] = m; __syncthreads();
  for (int s = 128; s > 0; s >>= 1) { if (tid < s) red[tid] = fmaxf(red[tid], red[tid+s]); __syncthreads(); }
  const float mx = red[0]; __syncthreads();
  float sum = 0.0f;
  for (int k = tid; k <= T; k += 256) { const float e = expf(sc[k] - mx); sc[k] = e; sum += e; }
  __syncthreads();
  red[tid] = sum; __syncthreads();
  for (int s = 128; s > 0; s >>= 1) { if (tid < s) red[tid] += red[tid+s]; __syncthreads(); }
  const float inv = 1.0f / red[0];
  const size_t base = ((size_t)b * T + i) * T;
  for (int k = tid; k < T; k += 256) {
    const float pl = (k <= i)    ? sc[k] * inv     : 0.0f;
    const float pr = (k + 1 > i) ? sc[k + 1] * inv : 0.0f;
    Pl[base + k] = (bf16)pl;
    Pr[base + k] = (bf16)pr;
  }
}

// outT[b][h][t] = bf16(out[t][b][h]); LDS-tiled 32x32, grid (T/32, H/32, B)
__global__ void transpose_cvt_kernel(const float* __restrict__ out, bf16* __restrict__ outT,
                                     int T, int B, int H) {
  __shared__ float tile[32][33];
  const int t0 = blockIdx.x * 32, h0 = blockIdx.y * 32, b = blockIdx.z;
  for (int r = threadIdx.y; r < 32; r += 8)
    tile[r][threadIdx.x] = out[((size_t)(t0 + r) * B + b) * H + h0 + threadIdx.x];
  __syncthreads();
  for (int r = threadIdx.y; r < 32; r += 8)
    outT[((size_t)b * H + h0 + r) * T + t0 + threadIdx.x] = (bf16)tile[threadIdx.x][r];
}

// h3[(t*B+b)][0:H]=outL, [H:2H]=outR, [2H:3H]=cv[b][t][:]
__global__ void h3_assemble_kernel(const float* __restrict__ outL, const float* __restrict__ outR,
                                   const float* __restrict__ cv, bf16* __restrict__ h3,
                                   int T, int B, int H) {
  const size_t n = (size_t)T * B * 3 * H;
  size_t idx = (size_t)blockIdx.x * blockDim.x + threadIdx.x;
  const size_t stride = (size_t)gridDim.x * blockDim.x;
  for (; idx < n; idx += stride) {
    const size_t row = idx / (3 * H);
    const int j = (int)(idx % (3 * H));
    const int t = (int)(row / B), b = (int)(row % B);
    float v;
    if (j < H)          v = outL[row * H + j];
    else if (j < 2 * H) v = outR[row * H + (j - H)];
    else                v = cv[((size_t)b * T + t) * H + (j - 2 * H)];
    h3[idx] = (bf16)v;
  }
}

// ---------------------------------------------------------------------------
// Orchestration
// ---------------------------------------------------------------------------
extern "C" void kernel_launch(void* const* d_in, const int* in_sizes, int n_in,
                              void* d_out, int out_size, void* d_ws, size_t ws_size,
                              hipStream_t stream) {
  (void)in_sizes; (void)n_in; (void)out_size; (void)ws_size;
  const int T = 256, B = 32, H = 1024, E = 1024, V = 32000, L = 2;
  const int KSPLIT = 4;   // split-K for the small recurrent GEMMs

  const int*   data_in[2] = { (const int*)d_in[0],  (const int*)d_in[1]  };
  const float* h0_in[2]   = { (const float*)d_in[2], (const float*)d_in[4] };
  const float* c0_in[2]   = { (const float*)d_in[3], (const float*)d_in[5] };
  const float* emb        =   (const float*)d_in[6];
  const float* Wih_in[2]  = { (const float*)d_in[7],  (const float*)d_in[11] };
  const float* Whh_in[2]  = { (const float*)d_in[8],  (const float*)d_in[12] };
  const float* bih_in[2]  = { (const float*)d_in[9],  (const float*)d_in[13] };
  const float* bhh_in[2]  = { (const float*)d_in[10], (const float*)d_in[14] };
  const float* attn_w     =   (const float*)d_in[15];
  const float* attn_b     =   (const float*)d_in[16];
  const float* trans_w    =   (const float*)d_in[17];
  const float* trans_b    =   (const float*)d_in[18];
  const float* dec_w      =   (const float*)d_in[19];
  const float* dec_b      =   (const float*)d_in[20];
  float* out = (float*)d_out;

  // ---- workspace carving (256B aligned) -----------------------------------
  char* ws = (char*)d_ws;
  size_t off = 0;
  auto alloc = [&](size_t bytes) -> char* {
    char* p = ws + off;
    off = (off + bytes + 255) & ~(size_t)255;
    return p;
  };
  // persistent region
  bf16* Wih_bf[2], *Whh_bf[2];
  for (int d = 0; d < 2; ++d) {
    Wih_bf[d] = (bf16*)alloc((size_t)L * 4 * H * H * sizeof(bf16));
    Whh_bf[d] = (bf16*)alloc((size_t)L * 4 * H * H * sizeof(bf16));
  }
  bf16*  transw_bf = (bf16*)alloc((size_t)H * 3 * H * sizeof(bf16));
  bf16*  decw_bf   = (bf16*)alloc((size_t)V * H * sizeof(bf16));
  float* outL      = (float*)alloc((size_t)T * B * H * sizeof(float));
  float* outR      = (float*)alloc((size_t)T * B * H * sizeof(float));
  float* a_s       = (float*)alloc((size_t)T * B * sizeof(float));
  float* sL_s      = (float*)alloc((size_t)T * B * sizeof(float));
  float* sR_s      = (float*)alloc((size_t)T * B * sizeof(float));
  bf16*  h_bf      = (bf16*)alloc((size_t)B * H * sizeof(bf16));
  float* c_st      = (float*)alloc((size_t)B * H * sizeof(float));

  const size_t phaseB = off;
  // phase B.1: LSTM transient
  bf16*  Xbf  = (bf16*)alloc((size_t)T * B * E * sizeof(bf16));
  bf16*  X1bf = (bf16*)alloc((size_t)T * B * H * sizeof(bf16));
  float* G    = (float*)alloc((size_t)T * B * 4 * H * sizeof(float));
  float* hW4  = (float*)alloc((size_t)KSPLIT * B * 4 * H * sizeof(float));
  // phase B.2: attention/decode transient (aliases B.1 — strictly later in stream)
  off = phaseB;
  bf16*  Pl      = (bf16*)alloc((size_t)B * T * T * sizeof(bf16));
  bf16*  Pr      = (bf16*)alloc((size_t)B * T * T * sizeof(bf16));
  bf16*  outLT   = (bf16*)alloc((size_t)B * H * T * sizeof(bf16));
  bf16*  outRT   = (bf16*)alloc((size_t)B * H * T * sizeof(bf16));
  float* cv      = (float*)alloc((size_t)B * T * H * sizeof(float));
  bf16*  h3      = (bf16*)alloc((size_t)T * B * 3 * H * sizeof(bf16));
  bf16*  pred_bf = (bf16*)alloc((size_t)T * B * H * sizeof(bf16));

  auto gemm = [&](const bf16* A, const bf16* Bm, void* C, int M, int N, int K,
                  long long sA, long long sB, long long sC, int batch,
                  const float* bias, int mode, int ksplit) {
    dim3 grid(N / 64, (M + BLOCK_ROWS - 1) / BLOCK_ROWS, batch * ksplit);
    switch (mode) {
      case 0: wmma_gemm_bf16_kernel<0><<<grid, 128, 0, stream>>>(A, Bm, (float*)C, M, N, K, sA, sB, sC, bias, ksplit); break;
      case 1: wmma_gemm_bf16_kernel<1><<<grid, 128, 0, stream>>>(A, Bm, (float*)C, M, N, K, sA, sB, sC, bias, ksplit); break;
      case 2: wmma_gemm_bf16_kernel<2><<<grid, 128, 0, stream>>>(A, Bm, (float*)C, M, N, K, sA, sB, sC, bias, ksplit); break;
      default: wmma_gemm_bf16_kernel<4><<<grid, 128, 0, stream>>>(A, Bm, (float*)C, M, N, K, sA, sB, sC, bias, ksplit); break;
    }
  };
  auto cvt = [&](const float* s, bf16* d, size_t n) {
    size_t nb = n / (256 * 8); if (nb == 0) nb = 1; if (nb > 4096) nb = 4096;
    cvt_bf16_kernel<<<(unsigned)nb, 256, 0, stream>>>(s, d, n);
  };

  // ---- weights -> bf16 -----------------------------------------------------
  for (int d = 0; d < 2; ++d) {
    cvt(Wih_in[d], Wih_bf[d], (size_t)L * 4 * H * H);
    cvt(Whh_in[d], Whh_bf[d], (size_t)L * 4 * H * H);
  }
  cvt(trans_w, transw_bf, (size_t)H * 3 * H);
  cvt(dec_w, decw_bf, (size_t)V * H);

  // ---- bidirectional 2-layer LSTM -----------------------------------------
  float* outArr[2] = { outL, outR };
  for (int dir = 0; dir < 2; ++dir) {
    embed_kernel<<<T * B, 256, 0, stream>>>(data_in[dir], emb, Xbf, T, B, E, dir);
    for (int l = 0; l < L; ++l) {
      const bf16* Xin = (l == 0) ? Xbf : X1bf;
      init_state_kernel<<<(B * H + 255) / 256, 256, 0, stream>>>(
          h0_in[dir] + (size_t)l * B * H, c0_in[dir] + (size_t)l * B * H, h_bf, c_st, B * H);
      // hoisted input GEMM over all timesteps: G = X @ Wih[l]^T  (8192 x 4096 x 1024)
      gemm(Xin, Wih_bf[dir] + (size_t)l * 4 * H * H, G, T * B, 4 * H, H,
           0, 0, 0, 1, nullptr, 0, 1);
      for (int t = 0; t < T; ++t) {
        // recurrent GEMM, split-K x4: hW4[p] = h @ Whh[l][:, pK:..]^T  (32x4096x256 each)
        gemm(h_bf, Whh_bf[dir] + (size_t)l * 4 * H * H, hW4, B, 4 * H, H,
             0, 0, 0, 1, nullptr, 0, KSPLIT);
        const int out_t = (dir == 1) ? (T - 1 - t) : t;   // right stack is re-reversed
        lstm_cell_kernel<<<(B * H + 255) / 256, 256, 0, stream>>>(
            G + (size_t)t * B * 4 * H, hW4,
            bih_in[dir] + (size_t)l * 4 * H, bhh_in[dir] + (size_t)l * 4 * H,
            c_st, h_bf,
            (l == L - 1) ? outArr[dir] : nullptr,
            (l == L - 1) ? nullptr : X1bf,
            B, H, t, out_t, KSPLIT);
      }
    }
  }

  // ---- attention -----------------------------------------------------------
  attn_dots_kernel<<<T * B, 256, 0, stream>>>(outL, outR, attn_w, attn_b, a_s, sL_s, sR_s, H);
  attn_softmax_kernel<<<dim3(T, B), 256, 0, stream>>>(a_s, sL_s, sR_s, Pl, Pr, T, B);
  transpose_cvt_kernel<<<dim3(T / 32, H / 32, B), dim3(32, 8), 0, stream>>>(outL, outLT, T, B, H);
  transpose_cvt_kernel<<<dim3(T / 32, H / 32, B), dim3(32, 8), 0, stream>>>(outR, outRT, T, B, H);
  // cv[b] = Pl[b] @ outL[b] + Pr[b] @ outR[b]   (batched 256x1024x256 WMMA GEMMs)
  gemm(Pl, outLT, cv, T, H, T, (long long)T * T, (long long)H * T, (long long)T * H,
       B, nullptr, 0, 1);
  gemm(Pr, outRT, cv, T, H, T, (long long)T * T, (long long)H * T, (long long)T * H,
       B, nullptr, 1, 1);

  // ---- output head ---------------------------------------------------------
  h3_assemble_kernel<<<4096, 256, 0, stream>>>(outL, outR, cv, h3, T, B, H);
  // pred = tanh(h3 @ trans_w^T + trans_b) fused to bf16   (8192 x 1024 x 3072)
  gemm(h3, transw_bf, pred_bf, T * B, H, 3 * H, 0, 0, 0, 1, trans_b, 2, 1);
  // decoded = pred @ dec_w^T + dec_b (streaming NT stores) (8192 x 32000 x 1024)
  gemm(pred_bf, decw_bf, out, T * B, V, H, 0, 0, 0, 1, dec_b, 4, 1);
}
